// LateJoinSAGE_14723147891049
// MI455X (gfx1250) — matrix operand order repaired and checked by reading, not compiled, not run on hardware
//
#include <hip/hip_runtime.h>

// ---------------------------------------------------------------------------
// CDNA5 (gfx1250) GraphSAGE forward.
// Wave32. Dense transforms use V_WMMA_F32_16X16X32_F16 (f16 in, f32 acc).
// Weights pre-packed into WMMA B-fragment layout (contiguous 32B per lane).
// Each wave computes 16 x (NT*16) output, reusing one A fragment NT times.
// Edge aggregation via global f32 atomics (L2-resident: N*160*4B = 64MB << 192MB L2).
// ---------------------------------------------------------------------------

typedef __attribute__((ext_vector_type(16))) _Float16 v16h;
typedef __attribute__((ext_vector_type(8)))  float    v8f;

#define LEAKY_SLOPE 0.01f

// ---- A fragment loader (ISA 7.12.2, 16-bit A 16x32, wave32) ---------------
// lane<16: row=lane, K = {0..7, 16..23}; lane>=16: row=lane-16, K = {8..15, 24..31}.
// Each lane's 16 halves = two contiguous 8-float runs -> 4 aligned float4 loads.
__device__ __forceinline__ v16h load_a_frag(const float* __restrict__ A, int row0,
                                            int k0, int lda, float s) {
  const int lane = threadIdx.x & 31;
  const int r    = lane & 15;
  const int kh   = lane >> 4;
  const float* arow = A + (size_t)(row0 + r) * lda + k0 + (kh << 3);
  const float4 p0 = *(const float4*)(arow + 0);
  const float4 p1 = *(const float4*)(arow + 4);
  const float4 p2 = *(const float4*)(arow + 16);
  const float4 p3 = *(const float4*)(arow + 20);
  __builtin_prefetch(arow + 32, 0, 1);           // next K-tile (global_prefetch_b8)
  v16h a;
  a[0] = (_Float16)(p0.x * s); a[1] = (_Float16)(p0.y * s);
  a[2] = (_Float16)(p0.z * s); a[3] = (_Float16)(p0.w * s);
  a[4] = (_Float16)(p1.x * s); a[5] = (_Float16)(p1.y * s);
  a[6] = (_Float16)(p1.z * s); a[7] = (_Float16)(p1.w * s);
  a[8]  = (_Float16)(p2.x * s); a[9]  = (_Float16)(p2.y * s);
  a[10] = (_Float16)(p2.z * s); a[11] = (_Float16)(p2.w * s);
  a[12] = (_Float16)(p3.x * s); a[13] = (_Float16)(p3.y * s);
  a[14] = (_Float16)(p3.z * s); a[15] = (_Float16)(p3.w * s);
  return a;
}

// ---- Multi-tile dual-input WMMA GEMM --------------------------------------
// C = A1@W1 [+ A2@W2] + bias, optional leaky-relu.
// W1/W2 are pre-packed fragments: [Kp/32][Np/16][lane:32][e:16] f16.
// One wave -> 16 rows x NT*16 cols. Requires M%16==0, Kp%32==0, (Np/16)%NT==0.
template <int NT>
__global__ void wmma_gemm(const float* __restrict__ A1, const _Float16* __restrict__ W1,
                          const float* __restrict__ A2, const _Float16* __restrict__ W2,
                          const float* __restrict__ bias,
                          const float* __restrict__ rowscale1,
                          float* __restrict__ C,
                          int M, int Kp, int Np, int mode /*0=none,1=leaky*/) {
  const int wave   = blockIdx.x * (blockDim.x >> 5) + (threadIdx.x >> 5);
  const int ntileN = Np >> 4;
  const int groups = ntileN / NT;
  const int total  = (M >> 4) * groups;
  if (wave >= total) return;                     // wave-uniform (EXEC all-1 for WMMA)
  const int row0 = (wave / groups) << 4;
  const int cg   = wave % groups;
  const int ct0  = cg * NT;
  const int lane = threadIdx.x & 31;

  v8f acc[NT];
#pragma unroll
  for (int t = 0; t < NT; ++t) acc[t] = (v8f){};

  const v16h* __restrict__ w1f = (const v16h*)W1;
  const int kt_n = Kp >> 5;

  {
    const int r = lane & 15;
    const float s = rowscale1 ? rowscale1[row0 + r] : 1.0f;
    for (int kt = 0; kt < kt_n; ++kt) {
      const v16h a = load_a_frag(A1, row0, kt << 5, Kp, s);
#pragma unroll
      for (int t = 0; t < NT; ++t) {
        const v16h b = w1f[((size_t)kt * ntileN + ct0 + t) * 32 + lane];
        acc[t] = __builtin_amdgcn_wmma_f32_16x16x32_f16(false, a, false, b,
                                                        (short)0, acc[t], false, false);
      }
    }
  }
  if (A2) {
    const v16h* __restrict__ w2f = (const v16h*)W2;
    for (int kt = 0; kt < kt_n; ++kt) {
      const v16h a = load_a_frag(A2, row0, kt << 5, Kp, 1.0f);
#pragma unroll
      for (int t = 0; t < NT; ++t) {
        const v16h b = w2f[((size_t)kt * ntileN + ct0 + t) * 32 + lane];
        acc[t] = __builtin_amdgcn_wmma_f32_16x16x32_f16(false, a, false, b,
                                                        (short)0, acc[t], false, false);
      }
    }
  }

  // C/D layout: 8 VGPRs; lanes 0-15 -> M=vgpr, lanes 16-31 -> M=8+vgpr; N=lane&15.
  const int c  = lane & 15;
  const int rb = (lane >> 4) << 3;
#pragma unroll
  for (int t = 0; t < NT; ++t) {
    const int col = ((ct0 + t) << 4) + c;
    const float bv = bias ? bias[col] : 0.0f;
#pragma unroll
    for (int r = 0; r < 8; ++r) {
      float v = acc[t][r] + bv;
      if (mode == 1) v = (v > 0.0f) ? v : LEAKY_SLOPE * v;
      C[(size_t)(row0 + rb + r) * Np + col] = v;
    }
  }
}

// ---- Weight packing: f32 (K x Nout, ld=Nout, +rowOff) -> fragment-major f16.
// dst logical dims: [Kp/32][Np/16][lane:32][e:16]; k = kt*32 + (lane>>4)*8 + (e&7) + (e>>3)*16,
// c = ct*16 + (lane&15). Zero-padded.
__global__ void pack_weight(const float* __restrict__ src, _Float16* __restrict__ dst,
                            int K, int Nout, int Kp, int Np, int rowOff) {
  const int ntN = Np >> 4;
  size_t i = (size_t)blockIdx.x * blockDim.x + threadIdx.x;
  const size_t st = (size_t)gridDim.x * blockDim.x;
  const size_t tot = (size_t)Kp * Np;
  for (; i < tot; i += st) {
    const int e    = (int)(i & 15);
    const int lane = (int)((i >> 4) & 31);
    const int ct   = (int)((i >> 9) % ntN);
    const int kt   = (int)((i >> 9) / ntN);
    const int k = (kt << 5) + ((lane >> 4) << 3) + (e & 7) + ((e >> 3) << 4);
    const int c = (ct << 4) + (lane & 15);
    const float v = (k < K && c < Nout) ? src[(size_t)(rowOff + k) * Nout + c] : 0.0f;
    dst[i] = (_Float16)v;
  }
}

// ---- Elementwise / graph kernels ------------------------------------------
__global__ void zero_f32(float* __restrict__ p, size_t n) {
  size_t i = (size_t)blockIdx.x * blockDim.x + threadIdx.x;
  const size_t st = (size_t)gridDim.x * blockDim.x;
  for (; i < n; i += st) p[i] = 0.0f;
}

__global__ void degree_kernel(const int* __restrict__ ei, float* __restrict__ deg, int E) {
  size_t i = (size_t)blockIdx.x * blockDim.x + threadIdx.x;
  const size_t st = (size_t)gridDim.x * blockDim.x;
  const size_t tot = (size_t)2 * E;
  for (; i < tot; i += st) {
    const int e = (i < (size_t)E) ? (int)i : (int)(i - E);
    const int dst = (i < (size_t)E) ? ei[2 * e + 1] : ei[2 * e];
    atomicAdd(&deg[dst], 1.0f);
  }
}

__global__ void invdeg_kernel(float* __restrict__ deg, int N) {
  size_t i = (size_t)blockIdx.x * blockDim.x + threadIdx.x;
  const size_t st = (size_t)gridDim.x * blockDim.x;
  for (; i < (size_t)N; i += st) deg[i] = 1.0f / fmaxf(deg[i], 1.0f);
}

// x0 = [node_feat | op_emb[opcode] | zero-pad] -> N x Kp
__global__ void build_x0(const float* __restrict__ nf, const int* __restrict__ opc,
                         const float* __restrict__ emb, float* __restrict__ X,
                         int N, int Kp) {
  size_t i = (size_t)blockIdx.x * blockDim.x + threadIdx.x;
  const size_t st = (size_t)gridDim.x * blockDim.x;
  const size_t tot = (size_t)N * Kp;
  for (; i < tot; i += st) {
    const int n = (int)(i / Kp), c = (int)(i % Kp);
    float v = 0.0f;
    if (c < 140)      v = nf[(size_t)n * 140 + c];
    else if (c < 148) v = emb[(size_t)opc[n] * 8 + (c - 140)];
    X[i] = v;
  }
}

__global__ void pad_bias(const float* __restrict__ src, float* __restrict__ dst,
                         int n, int np) {
  size_t i = (size_t)blockIdx.x * blockDim.x + threadIdx.x;
  const size_t st = (size_t)gridDim.x * blockDim.x;
  for (; i < (size_t)np; i += st) dst[i] = (i < (size_t)n) ? src[i] : 0.0f;
}

// AGG[dst] += X[src] over bidirectional edges, float4 granularity.
__global__ void scatter_kernel(const int* __restrict__ ei, const float* __restrict__ X,
                               float* __restrict__ AGG, int E, int Kp) {
  const int chunks = Kp >> 2;
  size_t i = (size_t)blockIdx.x * blockDim.x + threadIdx.x;
  const size_t st = (size_t)gridDim.x * blockDim.x;
  const size_t tot = (size_t)2 * E * chunks;
  for (; i < tot; i += st) {
    const int e  = (int)(i / chunks);
    const int ch = (int)(i % chunks);
    int s, d;
    if (e < E) { s = ei[2 * e];            d = ei[2 * e + 1]; }
    else       { const int e2 = e - E; s = ei[2 * e2 + 1]; d = ei[2 * e2]; }
    const float4 v = *(const float4*)(X + (size_t)s * Kp + (size_t)ch * 4);
    float* dp = AGG + (size_t)d * Kp + (size_t)ch * 4;
    atomicAdd(dp + 0, v.x); atomicAdd(dp + 1, v.y);
    atomicAdd(dp + 2, v.z); atomicAdd(dp + 3, v.w);
  }
}

// Row-wise L2 normalize (one wave per row), h /= max(||h||, 1e-12).
__global__ void l2norm_kernel(float* __restrict__ H, int M, int Np) {
  const int row = blockIdx.x * (blockDim.x >> 5) + (threadIdx.x >> 5);
  if (row >= M) return;
  const int lane = threadIdx.x & 31;
  float* hp = H + (size_t)row * Np;
  float ss = 0.0f;
  for (int c = lane; c < Np; c += 32) { const float v = hp[c]; ss += v * v; }
#pragma unroll
  for (int o = 16; o; o >>= 1) ss += __shfl_xor(ss, o, 32);
  const float inv = 1.0f / fmaxf(sqrtf(ss), 1e-12f);
  for (int c = lane; c < Np; c += 32) hp[c] *= inv;
}

// global_add_pool over batch ids (x is N x 64)
__global__ void pool_kernel(const float* __restrict__ X, const int* __restrict__ batch,
                            float* __restrict__ g, int N) {
  size_t i = (size_t)blockIdx.x * blockDim.x + threadIdx.x;
  const size_t st = (size_t)gridDim.x * blockDim.x;
  const size_t tot = (size_t)N * 64;
  for (; i < tot; i += st) {
    const int n = (int)(i >> 6), c = (int)(i & 63);
    atomicAdd(&g[(size_t)batch[n] * 64 + c], X[i]);
  }
}

// A_post = [g[row/cfg] (64) | config_feat (24) | pad (8)] -> R x 96
__global__ void build_post_kernel(const float* __restrict__ g, const float* __restrict__ cfg,
                                  float* __restrict__ A, int R, int cfgper) {
  size_t i = (size_t)blockIdx.x * blockDim.x + threadIdx.x;
  const size_t st = (size_t)gridDim.x * blockDim.x;
  const size_t tot = (size_t)R * 96;
  for (; i < tot; i += st) {
    const int r = (int)(i / 96), c = (int)(i % 96);
    float v = 0.0f;
    if (c < 64)      v = g[(size_t)(r / cfgper) * 64 + c];
    else if (c < 88) v = cfg[(size_t)r * 24 + (c - 64)];
    A[i] = v;
  }
}

// out[r] = dot(Z[r, 0:64], w2) + b2 (one wave per row)
__global__ void final_dot_kernel(const float* __restrict__ Z, const float* __restrict__ w2,
                                 const float* __restrict__ b2, float* __restrict__ out, int R) {
  const int row = blockIdx.x * (blockDim.x >> 5) + (threadIdx.x >> 5);
  if (row >= R) return;
  const int lane = threadIdx.x & 31;
  const float* z = Z + (size_t)row * 64;
  float s = z[lane] * w2[lane] + z[lane + 32] * w2[lane + 32];
#pragma unroll
  for (int o = 16; o; o >>= 1) s += __shfl_xor(s, o, 32);
  if (lane == 0) out[row] = s + b2[0];
}

// ---------------------------------------------------------------------------
extern "C" void kernel_launch(void* const* d_in, const int* in_sizes, int n_in,
                              void* d_out, int out_size, void* d_ws, size_t ws_size,
                              hipStream_t stream) {
  (void)n_in; (void)out_size; (void)ws_size;

  const float* node_feat   = (const float*)d_in[0];
  const int*   node_opcode = (const int*)  d_in[1];
  const int*   edge_index  = (const int*)  d_in[2];
  const float* config_feat = (const float*)d_in[3];
  const int*   batch       = (const int*)  d_in[5];
  const float* op_emb      = (const float*)d_in[6];
  const float* in_wl[3] = {(const float*)d_in[7],  (const float*)d_in[12], (const float*)d_in[17]};
  const float* in_bl[3] = {(const float*)d_in[8],  (const float*)d_in[13], (const float*)d_in[18]};
  const float* in_wr[3] = {(const float*)d_in[9],  (const float*)d_in[14], (const float*)d_in[19]};
  const float* in_lw[3] = {(const float*)d_in[10], (const float*)d_in[15], (const float*)d_in[20]};
  const float* in_lb[3] = {(const float*)d_in[11], (const float*)d_in[16], (const float*)d_in[21]};
  const float* post_w1 = (const float*)d_in[22];
  const float* post_b1 = (const float*)d_in[23];
  const float* post_w2 = (const float*)d_in[24];
  const float* post_b2 = (const float*)d_in[25];
  float* out = (float*)d_out;

  const int N  = in_sizes[0] / 140;
  const int E  = in_sizes[2] / 2;
  const int BC = in_sizes[3] / 24;
  const int B  = in_sizes[4];
  const int CFGper = BC / B;

  const int Kp[3]    = {160, 64, 64};   // padded layer in-dims (K %32 == 0)
  const int Kreal[3] = {148, 64, 64};

  // ---- carve workspace ----
  size_t off = 0;
  char* ws = (char*)d_ws;
  auto carve = [&](size_t bytes) -> char* {
    char* p = ws + off;
    off += (bytes + 255) & ~(size_t)255;
    return p;
  };
  float* deg = (float*)carve((size_t)N * 4);
  float* XA  = (float*)carve((size_t)N * 160 * 4);
  float* XB  = (float*)carve((size_t)N * 160 * 4);
  float* Hb  = (float*)carve((size_t)N * 160 * 4);
  _Float16 *WL[3], *WR[3], *LWT[3], *LWB[3];
  float *BL[3], *LB[3];
  for (int l = 0; l < 3; ++l) {
    WL[l]  = (_Float16*)carve((size_t)Kp[l] * Kp[l] * 2);
    WR[l]  = (_Float16*)carve((size_t)Kp[l] * Kp[l] * 2);
    LWT[l] = (_Float16*)carve((size_t)Kp[l] * 64 * 2);
    LWB[l] = (_Float16*)carve((size_t)Kp[l] * 64 * 2);
    BL[l]  = (float*)carve((size_t)Kp[l] * 4);
    LB[l]  = (float*)carve(64 * 4);
  }
  _Float16* PW1 = (_Float16*)carve((size_t)96 * 64 * 2);
  float* PB1    = (float*)carve(64 * 4);
  float* GPOOL  = (float*)carve((size_t)B * 64 * 4);
  float* APOST  = (float*)carve((size_t)BC * 96 * 4);
  float* ZPOST  = (float*)carve((size_t)BC * 64 * 4);

  auto gsb = [](size_t work, int bs) -> int {
    size_t b = (work + bs - 1) / (size_t)bs;
    if (b > (size_t)1 << 18) b = (size_t)1 << 18;
    if (b < 1) b = 1;
    return (int)b;
  };
  // 4 waves (128 threads) per block; each wave does a 16 x NT*16 output strip.
  auto gemm_blocks = [](int M, int Np, int NT) -> int {
    const int waves = (M >> 4) * ((Np >> 4) / NT);
    return (waves + 3) / 4;
  };

  // ---- degrees -> inv_deg ----
  zero_f32<<<gsb(N, 256), 256, 0, stream>>>(deg, (size_t)N);
  degree_kernel<<<gsb((size_t)2 * E, 256), 256, 0, stream>>>(edge_index, deg, E);
  invdeg_kernel<<<gsb(N, 256), 256, 0, stream>>>(deg, N);

  // ---- x0 = concat(node_feat, op_emb[opcode]) ----
  build_x0<<<gsb((size_t)N * 160, 256), 256, 0, stream>>>(node_feat, node_opcode,
                                                          op_emb, XA, N, 160);

  // ---- weight packing to WMMA fragment layout (f16) ----
  for (int l = 0; l < 3; ++l) {
    const size_t wsz = (size_t)Kp[l] * Kp[l];
    pack_weight<<<gsb(wsz, 256), 256, 0, stream>>>(in_wl[l], WL[l], Kreal[l], Kreal[l], Kp[l], Kp[l], 0);
    pack_weight<<<gsb(wsz, 256), 256, 0, stream>>>(in_wr[l], WR[l], Kreal[l], Kreal[l], Kp[l], Kp[l], 0);
    const size_t lsz = (size_t)Kp[l] * 64;
    pack_weight<<<gsb(lsz, 256), 256, 0, stream>>>(in_lw[l], LWT[l], Kreal[l], 64, Kp[l], 64, 0);
    pack_weight<<<gsb(lsz, 256), 256, 0, stream>>>(in_lw[l], LWB[l], Kreal[l], 64, Kp[l], 64, Kreal[l]);
    pad_bias<<<1, 256, 0, stream>>>(in_bl[l], BL[l], Kreal[l], Kp[l]);
    pad_bias<<<1, 256, 0, stream>>>(in_lb[l], LB[l], 64, 64);
  }
  pack_weight<<<gsb(96 * 64, 256), 256, 0, stream>>>(post_w1, PW1, 88, 64, 96, 64, 0);
  pad_bias<<<1, 256, 0, stream>>>(post_b1, PB1, 64, 64);

  // ---- 3 SAGE layers ----
  float* X  = XA;
  float* XN = XB;
  for (int l = 0; l < 3; ++l) {
    const int K = Kp[l];
    zero_f32<<<gsb((size_t)N * K, 256), 256, 0, stream>>>(XN, (size_t)N * K);
    scatter_kernel<<<gsb((size_t)2 * E * (K >> 2), 256), 256, 0, stream>>>(edge_index, X, XN, E, K);
    // h = (agg * inv_deg) @ wl + x @ wr + bl
    if (K == 160) {
      wmma_gemm<5><<<gemm_blocks(N, K, 5), 128, 0, stream>>>(XN, WL[l], X, WR[l],
                                                             BL[l], deg, Hb, N, K, K, 0);
    } else {
      wmma_gemm<4><<<gemm_blocks(N, K, 4), 128, 0, stream>>>(XN, WL[l], X, WR[l],
                                                             BL[l], deg, Hb, N, K, K, 0);
    }
    l2norm_kernel<<<gsb((size_t)N * 32, 256), 256, 0, stream>>>(Hb, N, K);
    // x_next = leaky_relu([h | x] @ lw + lb) = leaky(h@lw_top + x@lw_bot + lb)
    wmma_gemm<4><<<gemm_blocks(N, 64, 4), 128, 0, stream>>>(Hb, LWT[l], X, LWB[l],
                                                            LB[l], nullptr, XN, N, K, 64, 1);
    float* t = X; X = XN; XN = t;
  }

  // ---- global add pool + head ----
  zero_f32<<<1, 512, 0, stream>>>(GPOOL, (size_t)B * 64);
  pool_kernel<<<gsb((size_t)N * 64, 256), 256, 0, stream>>>(X, batch, GPOOL, N);
  build_post_kernel<<<gsb((size_t)BC * 96, 256), 256, 0, stream>>>(GPOOL, config_feat,
                                                                   APOST, BC, CFGper);
  wmma_gemm<4><<<gemm_blocks(BC, 64, 4), 128, 0, stream>>>(APOST, PW1, nullptr, nullptr,
                                                           PB1, nullptr, ZPOST, BC, 96, 64, 1);
  final_dot_kernel<<<gsb((size_t)BC * 32, 256), 256, 0, stream>>>(ZPOST, post_w2, post_b2,
                                                                  out, BC);
}
